// NeuralAdditiveModel_60739427500189
// MI455X (gfx1250) — matrix operand order repaired and checked by reading, not compiled, not run on hardware
//
#include <hip/hip_runtime.h>
#include <hip/hip_bf16.h>
#include <stddef.h>

// NAM forward: B=16384, F=128, U1=64, H1=64, H2=32
// per-feature MLP mapped onto v_wmma_f32_16x16x32_f16 (f16 in, f32 accum).

typedef __attribute__((ext_vector_type(16))) _Float16 v16h;
typedef __attribute__((ext_vector_type(8)))  float    v8f;

#define NB 16384
#define NF 128
#define NU 64
#define NH1 64
#define NH2 32

#define BLK_WAVES 8
#define ROWS_PER_BLOCK (BLK_WAVES * 16)   // 128 rows per block pass
#define TILES_PER_BLOCK 8                 // 8 passes -> 1024 rows per block

__device__ __forceinline__ v8f wmma16x16x32(v16h a, v16h b, v8f c) {
    return __builtin_amdgcn_wmma_f32_16x16x32_f16(
        /*neg_a=*/false, a, /*neg_b=*/false, b,
        /*c_mod=*/(short)0, c, /*reuse_a=*/false, /*reuse_b=*/false);
}

__global__ __launch_bounds__(256) void nam_forward_kernel(
    const float* __restrict__ x,      // [B,F]
    const float* __restrict__ w1,     // [F,U1]
    const float* __restrict__ b1,     // [F,U1]
    const float* __restrict__ w2,     // [F,U1,H1]
    const float* __restrict__ b2,     // [F,H1]
    const float* __restrict__ w3,     // [F,H1,H2]
    const float* __restrict__ b3,     // [F,H2]
    const float* __restrict__ w_out,  // [F,H2]
    const float* __restrict__ b_out,  // [F]
    float* __restrict__ pfo)          // [B,F]
{
    // B-operand layout: element i of lane L = W[K][N],
    //   K = i + 16*(L>>4) (+32*kt), N = (L&15) + 16*nt
    __shared__ __attribute__((aligned(32))) _Float16 sW2[4][2][32][16]; // 8 KB
    __shared__ __attribute__((aligned(32))) _Float16 sW3[2][2][32][16]; // 4 KB
    __shared__ float sW1[NU], sB1[NU], sB2[NH1], sB3[NH2], sWout[NH2];
    __shared__ float sBout;
    __shared__ _Float16 sH2[BLK_WAVES][16][64];  // 16 KB, per-wave scratch
    __shared__ float    sH3[BLK_WAVES][16][32];  // 16 KB, per-wave scratch

    const int t = threadIdx.x;
    const int f = blockIdx.y;

    // ---- Stage weights into LDS (B-operand order, f32 -> f16) ----
    const size_t wbase2 = (size_t)f * NU * NH1;  // 4096
    const size_t wbase3 = (size_t)f * NH1 * NH2; // 2048
#pragma unroll
    for (int j = 0; j < 16; ++j) {               // 4096 elems / 256 thr
        int e  = t * 16 + j;
        int i  = e & 15;
        int ln = (e >> 4) & 31;
        int kt = (e >> 9) & 1;
        int nt = (e >> 10) & 3;
        int K  = (i & 15) + 16 * (ln >> 4) + 32 * kt;
        int N  = (ln & 15) + 16 * nt;
        sW2[nt][kt][ln][i] = (_Float16)w2[wbase2 + (size_t)K * NH1 + N];
    }
#pragma unroll
    for (int j = 0; j < 8; ++j) {                // 2048 elems / 256 thr
        int e  = t * 8 + j;
        int i  = e & 15;
        int ln = (e >> 4) & 31;
        int kt = (e >> 9) & 1;
        int nt = (e >> 10) & 1;
        int K  = (i & 15) + 16 * (ln >> 4) + 32 * kt;
        int N  = (ln & 15) + 16 * nt;
        sW3[nt][kt][ln][i] = (_Float16)w3[wbase3 + (size_t)K * NH2 + N];
    }
    if (t < NU)  { sW1[t] = w1[f * NU + t]; sB1[t] = b1[f * NU + t]; sB2[t] = b2[f * NH1 + t]; }
    if (t < NH2) { sB3[t] = b3[f * NH2 + t]; sWout[t] = w_out[f * NH2 + t]; }
    if (t == 0)  { sBout = b_out[f]; }
    __syncthreads();

    const int wave   = t >> 5;
    const int lane   = t & 31;
    const int laneM  = lane & 15;
    const int laneHi = lane >> 4;

    for (int it = 0; it < TILES_PER_BLOCK; ++it) {
        const int b0  = blockIdx.x * (ROWS_PER_BLOCK * TILES_PER_BLOCK)
                        + it * ROWS_PER_BLOCK + wave * 16;
        const int row = b0 + laneM;
        const float xv = x[(size_t)row * NF + f];
        if (it + 1 < TILES_PER_BLOCK)
            __builtin_prefetch(&x[(size_t)(row + ROWS_PER_BLOCK) * NF + f], 0, 3);

        // ---- Stage 1: h1 = relu(x*w1 + b1), built directly in A layout ----
        v16h a0, a1;
#pragma unroll
        for (int i = 0; i < 16; ++i) {
            int u = i + 16 * laneHi;   // K index 0..31
            float h0 = xv * sW1[u] + sB1[u];
            float h1v = xv * sW1[u + 32] + sB1[u + 32];
            a0[i] = (_Float16)(h0  > 0.f ? h0  : 0.f);
            a1[i] = (_Float16)(h1v > 0.f ? h1v : 0.f);
        }

        // ---- Stage 2: h2 = relu(h1 @ w2 + b2)   (M16 x K64 x N64) ----
#pragma unroll
        for (int nt = 0; nt < 4; ++nt) {
            v8f c = {};
            v16h bw0 = *(const v16h*)&sW2[nt][0][lane][0];
            c = wmma16x16x32(a0, bw0, c);
            v16h bw1 = *(const v16h*)&sW2[nt][1][lane][0];
            c = wmma16x16x32(a1, bw1, c);
            const float bias = sB2[laneM + 16 * nt];
#pragma unroll
            for (int v = 0; v < 8; ++v) {
                float hv = c[v] + bias;
                hv = hv > 0.f ? hv : 0.f;
                sH2[wave][v + 8 * laneHi][laneM + 16 * nt] = (_Float16)hv;
            }
        }
        asm volatile("s_wait_dscnt 0" ::: "memory");   // same-wave LDS store->load

        // ---- Stage 3: h3 = relu(h2 @ w3 + b3)   (M16 x K64 x N32) ----
        v16h e0, e1;
#pragma unroll
        for (int i = 0; i < 16; ++i) {
            int K = i + 16 * laneHi;
            e0[i] = sH2[wave][laneM][K];
            e1[i] = sH2[wave][laneM][K + 32];
        }
#pragma unroll
        for (int nt = 0; nt < 2; ++nt) {
            v8f c = {};
            v16h bw0 = *(const v16h*)&sW3[nt][0][lane][0];
            c = wmma16x16x32(e0, bw0, c);
            v16h bw1 = *(const v16h*)&sW3[nt][1][lane][0];
            c = wmma16x16x32(e1, bw1, c);
            const float bias = sB3[laneM + 16 * nt];
#pragma unroll
            for (int v = 0; v < 8; ++v) {
                float hv = c[v] + bias;
                sH3[wave][v + 8 * laneHi][laneM + 16 * nt] = hv > 0.f ? hv : 0.f;
            }
        }
        asm volatile("s_wait_dscnt 0" ::: "memory");

        // ---- Stage 4: per_feature_output = h3 . w_out + b_out ----
        if (lane < 16) {
            float s = sBout;
#pragma unroll
            for (int k = 0; k < NH2; ++k) s += sH3[wave][lane][k] * sWout[k];
            pfo[(size_t)(b0 + lane) * NF + f] = s;
        }
    }
}

// linear_combination[b] = intercept + sum_f pfo[b,f]; one wave per row.
__global__ __launch_bounds__(256) void nam_reduce_kernel(
    const float* __restrict__ pfo, const float* __restrict__ intercept,
    float* __restrict__ lin)
{
    const int wave = threadIdx.x >> 5;
    const int lane = threadIdx.x & 31;
    const int b = blockIdx.x * 8 + wave;
    const float* rowp = pfo + (size_t)b * NF;
    float s = rowp[lane] + rowp[lane + 32] + rowp[lane + 64] + rowp[lane + 96];
#pragma unroll
    for (int off = 16; off > 0; off >>= 1) s += __shfl_xor(s, off, 32);
    if (lane == 0) lin[b] = s + intercept[0];
}

extern "C" void kernel_launch(void* const* d_in, const int* in_sizes, int n_in,
                              void* d_out, int out_size, void* d_ws, size_t ws_size,
                              hipStream_t stream) {
    (void)in_sizes; (void)n_in; (void)out_size; (void)d_ws; (void)ws_size;
    const float* x         = (const float*)d_in[0];
    const float* w1        = (const float*)d_in[1];
    const float* b1        = (const float*)d_in[2];
    const float* w2        = (const float*)d_in[3];
    const float* b2        = (const float*)d_in[4];
    const float* w3        = (const float*)d_in[5];
    const float* b3        = (const float*)d_in[6];
    const float* w_out     = (const float*)d_in[7];
    const float* b_out     = (const float*)d_in[8];
    const float* intercept = (const float*)d_in[9];

    float* lin = (float*)d_out;        // [B]   first output
    float* pfo = lin + NB;             // [B,F] second output

    dim3 grid1(NB / (ROWS_PER_BLOCK * TILES_PER_BLOCK), NF);  // (16, 128)
    nam_forward_kernel<<<grid1, 256, 0, stream>>>(x, w1, b1, w2, b2, w3, b3,
                                                  w_out, b_out, pfo);
    nam_reduce_kernel<<<NB / 8, 256, 0, stream>>>(pfo, intercept, lin);
}
